// TimesBlock_67611375174283
// MI455X (gfx1250) — matrix operand order repaired and checked by reading, not compiled, not run on hardware
//
#include <hip/hip_runtime.h>
#include <hip/hip_bf16.h>

typedef __attribute__((ext_vector_type(16))) _Float16 v16h;
typedef __attribute__((ext_vector_type(8)))  float    v8f;
typedef __attribute__((ext_vector_type(4)))  unsigned int v4u;
typedef __attribute__((ext_vector_type(4)))  int v4i;
typedef __attribute__((address_space(1))) v4i as1_v4i;
typedef __attribute__((address_space(3))) v4i as3_v4i;

union ABFrag { v4u u[2]; v16h h; };
union H2 { _Float16 h[2]; unsigned int u; };
union H8 { _Float16 h[8]; v4u u; };

#define SEQ_T   384
#define TOTAL_L 480
#define CDIM    512
#define NHEADS  8
#define HDIM    64
#define TOPK    5
#define BATCH   32
#define NIMG    (BATCH*TOPK)      // 160
#define SMAX    768               // >= worst-case full_len (766)
#define KTOT    (CDIM*9)          // 4608
#define KSTEPS  (KTOT/32)         // 144

#if __has_builtin(__builtin_amdgcn_global_load_async_to_lds_b128)
#define HAVE_ASYNC_LDS 1
#else
#define HAVE_ASYNC_LDS 0
#endif

__device__ __forceinline__ void async_copy16(unsigned int* ldsdst, const unsigned int* gsrc) {
#if HAVE_ASYNC_LDS
    __builtin_amdgcn_global_load_async_to_lds_b128(
        (as1_v4i*)(void*)gsrc,
        (as3_v4i*)(void*)ldsdst, 0, 0);
#else
    *(v4u*)ldsdst = *(const v4u*)gsrc;      // sync fallback: global load + ds_store
#endif
}

__device__ __forceinline__ void wait_async_prev() {   // 8 newer issues outstanding
#if HAVE_ASYNC_LDS
#if __has_builtin(__builtin_amdgcn_s_wait_asynccnt)
    __builtin_amdgcn_s_wait_asynccnt(8);
#else
    asm volatile("s_wait_asynccnt 8" ::: "memory");
#endif
#endif
}

__device__ __forceinline__ void wait_async_all() {
#if HAVE_ASYNC_LDS
#if __has_builtin(__builtin_amdgcn_s_wait_asynccnt)
    __builtin_amdgcn_s_wait_asynccnt(0);
#else
    asm volatile("s_wait_asynccnt 0" ::: "memory");
#endif
#endif
}

// ---------------------------------------------------------------- meta init
__global__ void init_meta_kernel(int* meta) {
    if (threadIdx.x == 0) { meta[0] = 1; meta[1] = 0; meta[2] = 0; }
}

// ------------------------------------------- moving average / seasonal / out init
__global__ __launch_bounds__(256) void trend_kernel(
    const float* __restrict__ x, float* __restrict__ s,
    float* __restrict__ trend, float* __restrict__ outp) {
    int tid = blockIdx.x * 256 + threadIdx.x;      // 32*512 threads
    int b = tid >> 9, c = tid & 511;
    const float* xp = x + (size_t)b * SEQ_T * CDIM + c;
    float run = 0.f;
    for (int t = 0; t < 13; ++t) run += xp[(size_t)t * CDIM];
    float lastv = 0.f, lasttr = 0.f;
    for (int t = 0; t < SEQ_T; ++t) {
        int lo = t - 12; if (lo < 0) lo = 0;
        int hi = t + 12; if (hi > SEQ_T - 1) hi = SEQ_T - 1;
        float tr = run / (float)(hi - lo + 1);
        float xv = xp[(size_t)t * CDIM];
        size_t idx = ((size_t)b * SEQ_T + t) * CDIM + c;
        trend[idx] = tr;
        s[idx] = xv - tr;
        outp[((size_t)b * TOTAL_L + t) * CDIM + c] = xv + tr;
        if (t + 13 <= SEQ_T - 1) run += xp[(size_t)(t + 13) * CDIM];
        if (t - 12 >= 0)         run -= xp[(size_t)(t - 12) * CDIM];
        lastv = xv; lasttr = tr;
    }
    float tail = lastv + lasttr;
    for (int t = SEQ_T; t < TOTAL_L; ++t)
        outp[((size_t)b * TOTAL_L + t) * CDIM + c] = tail;
}

// ------------------------------------------------ circular autocorrelation
__global__ __launch_bounds__(384) void autocorr_kernel(
    const float* __restrict__ s, float* __restrict__ r) {
    __shared__ __align__(16) float lds[SEQ_T * 18];   // stride 18: conflict-free, 8B aligned
    int bh = blockIdx.x;
    int b = bh >> 3, h = bh & 7;
    int tau = threadIdx.x;
    float acc = 0.f;
    for (int dc = 0; dc < 4; ++dc) {                  // 4 chunks of 16 of hd=64
        const float* src = s + ((size_t)b * SEQ_T + tau) * CDIM + h * HDIM + dc * 16;
        #pragma unroll
        for (int d4 = 0; d4 < 4; ++d4) {
            float4 v = *(const float4*)(src + d4 * 4);
            float* dst = &lds[tau * 18 + d4 * 4];
            *(float2*)(dst)     = make_float2(v.x, v.y);
            *(float2*)(dst + 2) = make_float2(v.z, v.w);
        }
        __syncthreads();
        for (int t = 0; t < SEQ_T; ++t) {
            int t2 = t + tau; if (t2 >= SEQ_T) t2 -= SEQ_T;
            const float* ar = &lds[t * 18];
            const float* br = &lds[t2 * 18];
            #pragma unroll
            for (int d = 0; d < 16; d += 2) {
                float2 av = *(const float2*)(ar + d);
                float2 bv = *(const float2*)(br + d);
                acc = fmaf(av.x, bv.x, acc);
                acc = fmaf(av.y, bv.y, acc);
            }
        }
        __syncthreads();
    }
    r[(size_t)bh * SEQ_T + tau] = (tau == 0) ? 0.f : acc * (1.f / 64.f);
}

// -------------------------------------------------------- top-k + softmax
__global__ void topk_kernel(const float* __restrict__ r,
                            int* __restrict__ periods, float* __restrict__ weights,
                            int* __restrict__ meta) {
    int bh = threadIdx.x;                              // 1 block, 256 threads
    const float* row = r + (size_t)bh * SEQ_T;
    int idxs[TOPK]; float vals[TOPK];
    for (int kk = 0; kk < TOPK; ++kk) {
        float best = -3.4e38f; int bi = 0;
        for (int t = 0; t < SEQ_T; ++t) {
            bool used = false;
            for (int q = 0; q < kk; ++q) used |= (idxs[q] == t);
            float v = row[t];
            if (!used && v > best) { best = v; bi = t; }
        }
        idxs[kk] = bi; vals[kk] = best;
    }
    float m = vals[0];
    float sum = 0.f;
    float e[TOPK];
    for (int kk = 0; kk < TOPK; ++kk) { e[kk] = __expf(vals[kk] - m); sum += e[kk]; }
    int pml = 1;
    for (int kk = 0; kk < TOPK; ++kk) {
        periods[bh * TOPK + kk] = idxs[kk];
        weights[bh * TOPK + kk] = e[kk] / sum;
        if (idxs[kk] > pml) pml = idxs[kk];
    }
    atomicMax(&meta[0], pml);
}

// --------------------------------------- lag-weight means + derived shapes
__global__ void finalize_kernel(const float* __restrict__ weights,
                                float* __restrict__ wlag, int* __restrict__ meta) {
    int tid = threadIdx.x;
    if (tid < BATCH * TOPK) {
        int b = tid / TOPK, kk = tid % TOPK;
        float sum = 0.f;
        for (int h = 0; h < NHEADS; ++h) sum += weights[((b * NHEADS + h) * TOPK) + kk];
        wlag[tid] = sum * (1.f / NHEADS);
    }
    if (tid == BATCH * TOPK) {
        int p = meta[0]; if (p < 1) p = 1;
        int cyc = (TOTAL_L + p - 1) / p;
        meta[0] = p; meta[1] = cyc; meta[2] = cyc * p;   // full_len
    }
}

// ---------------------------------------- build rolled+padded f16 activations
__global__ __launch_bounds__(256) void roll_kernel(
    const float* __restrict__ s, const int* __restrict__ periods,
    const int* __restrict__ meta, _Float16* __restrict__ act1) {
    size_t gid = (size_t)blockIdx.x * 256 + threadIdx.x;   // NIMG*SMAX*64
    int cvec = (int)(gid & 63);
    size_t t1 = gid >> 6;
    int pos = (int)(t1 % SMAX);
    int n   = (int)(t1 / SMAX);
    int full_len = meta[2];
    if (pos >= full_len) return;
    int b = n / TOPK, kk = n % TOPK;
    int c0 = cvec * 8;
    int h = c0 >> 6;
    int p = periods[(b * NHEADS + h) * TOPK + kk];
    int l = pos < TOTAL_L ? pos : TOTAL_L - 1;
    int st = l + p; if (st >= TOTAL_L) st -= TOTAL_L;      // l<=479, p<=383 -> one sub enough
    if (st > SEQ_T - 1) st = SEQ_T - 1;                    // replicate time pad
    const float* src = s + ((size_t)(b * SEQ_T + st)) * CDIM + c0;
    float4 a = *(const float4*)(src);
    float4 bq = *(const float4*)(src + 4);
    H8 o;
    o.h[0] = (_Float16)a.x;  o.h[1] = (_Float16)a.y;
    o.h[2] = (_Float16)a.z;  o.h[3] = (_Float16)a.w;
    o.h[4] = (_Float16)bq.x; o.h[5] = (_Float16)bq.y;
    o.h[6] = (_Float16)bq.z; o.h[7] = (_Float16)bq.w;
    *(v4u*)(act1 + ((size_t)n * SMAX + pos) * CDIM + c0) = o.u;
}

// ------------- pre-swizzle weights into per-lane WMMA B-fragment layout (f16)
__global__ __launch_bounds__(256) void pack_kernel(
    const float* __restrict__ W, unsigned int* __restrict__ Bpack) {
    size_t e = (size_t)blockIdx.x * 256 + threadIdx.x;     // 9*16*32*32*8
    int v    = (int)(e & 7);
    int lane = (int)((e >> 3) & 31);
    int nt   = (int)((e >> 8) & 31);
    int cinc = (int)((e >> 13) & 15);
    int tap  = (int)(e >> 17);
    int rr = tap / 3, ss = tap - rr * 3;
    int ncol = nt * 16 + (lane & 15);
    int k0 = cinc * 32 + ((v & 4) ? 16 : 0) + ((lane & 16) ? 8 : 0) + 2 * (v & 3);
    float w0 = W[((((size_t)ncol * CDIM) + k0) * 3 + rr) * 3 + ss];
    float w1 = W[((((size_t)ncol * CDIM) + k0 + 1) * 3 + rr) * 3 + ss];
    H2 p; p.h[0] = (_Float16)w0; p.h[1] = (_Float16)w1;
    Bpack[e] = p.u;
}

// --------------------------------------------------- WMMA implicit-GEMM conv
// Double-buffered K pipeline: async global->LDS staging of step k+1 overlaps WMMA of step k.
// MODE 0: conv1 over 5 lags of batch b, fused bias+GELU, lag-weighted sum -> act2 (f16)
// MODE 1: conv2 over batch b, bias, += into d_out (pos < 480)
template <int MODE>
__global__ __launch_bounds__(256) void conv_kernel(
    const _Float16* __restrict__ actA, const unsigned int* __restrict__ Bpack,
    const float* __restrict__ bias, const float* __restrict__ wlag,
    const int* __restrict__ meta, _Float16* __restrict__ act2,
    float* __restrict__ outp) {
    extern __shared__ __align__(16) unsigned int smem[];
    unsigned int* Blds = smem;            // 2 x 8192 u32  (2 x 32KB)
    unsigned int* Alds = smem + 16384;    // 2 x 512 u32   (2 x 2KB)

    const int p_max    = meta[0];
    const int cycles   = meta[1];
    const int full_len = meta[2];
    const int mbase = blockIdx.x * 32;
    const int b = blockIdx.y;
    if (mbase >= full_len) return;

    const int tid  = threadIdx.x;
    const int lane = tid & 31;
    const int wv   = tid >> 5;
    const int mt   = wv >> 2;         // 0..1 : M sub-tile
    const int nq   = wv & 3;          // 0..3 : 128-wide N slice
    const int arow = mt * 16 + (lane & 15);
    const int asel = (lane & 16) ? 4 : 0;

    // im2col guard state, hoisted: division once per block
    const bool isA = (tid < 128);
    int sRow = 0, sSeg = 0, sPos = 0, sI = 0, sJ = 0;
    bool sIn = false;
    if (isA) {
        sRow = tid >> 2; sSeg = tid & 3;
        sPos = mbase + sRow;
        sIn  = sPos < full_len;
        sI   = sPos / p_max;
        sJ   = sPos - sI * p_max;
    }

    const v8f zero8 = {0.f,0.f,0.f,0.f,0.f,0.f,0.f,0.f};
    float res[8][8];
    if (MODE == 0) {
        #pragma unroll
        for (int j = 0; j < 8; ++j)
            #pragma unroll
            for (int vg = 0; vg < 8; ++vg) res[j][vg] = 0.f;
    }

    v8f acc[8];
    const int kkEnd = (MODE == 0) ? TOPK : 1;
    for (int kk = 0; kk < kkEnd; ++kk) {
        const int n_img = (MODE == 0) ? (b * TOPK + kk) : b;
        const _Float16* act = actA + (size_t)n_img * SMAX * CDIM;
        #pragma unroll
        for (int j = 0; j < 8; ++j) acc[j] = zero8;

        int curoff = -1;   // A-source offset (halves) for current tap, -1 = padded
        auto stage = [&](int ks, int bi) {
            // ---- A tile (im2col, 32 rows x 32 halves), tap guard updated every 16 steps
            if (isA) {
                int tapn = ks >> 4, cincn = ks & 15;
                if ((ks & 15) == 0) {
                    int r3 = tapn / 3;
                    int di = r3 - 1, dj = tapn - r3 * 3 - 1;
                    int ii = sI + di, jj = sJ + dj;
                    bool ok = sIn && (ii >= 0) && (ii < cycles) && (jj >= 0) && (jj < p_max);
                    curoff = ok ? (sPos + di * p_max + dj) * CDIM : -1;
                }
                v4u av = {0u, 0u, 0u, 0u};
                if (curoff >= 0)
                    av = *(const v4u*)(act + curoff + cincn * 32 + sSeg * 8);
                *(v4u*)&Alds[bi * 512 + sRow * 16 + sSeg * 4] = av;
            }
            // ---- B tile: 32K x 512N halves, async global->LDS (8 x b128 per thread)
            const unsigned int* bsrc = Bpack + ((size_t)ks << 13);
            unsigned int* bdst = &Blds[bi * 8192];
            #pragma unroll
            for (int q = 0; q < 8; ++q) {
                int idx = q * 1024 + tid * 4;
                async_copy16(&bdst[idx], &bsrc[idx]);
            }
        };

        stage(0, 0);                             // prologue
        for (int kc = 0; kc < KSTEPS; ++kc) {
            const int cur = kc & 1, nxt = cur ^ 1;
            const bool havenext = (kc + 1 < KSTEPS);
            if (havenext) { stage(kc + 1, nxt); wait_async_prev(); }
            else          { wait_async_all(); }
            __syncthreads();                     // buffer `cur` fully staged & visible

            ABFrag af;
            const unsigned int* abase = &Alds[cur * 512];
            af.u[0] = *(const v4u*)&abase[arow * 16 + asel];
            af.u[1] = *(const v4u*)&abase[arow * 16 + 8 + asel];
            const unsigned int* bbase = &Blds[cur * 8192];
            #pragma unroll
            for (int j = 0; j < 8; ++j) {
                int nt = nq * 8 + j;
                const unsigned int* bl = &bbase[((nt * 32) + lane) * 8];
                ABFrag bf;
                bf.u[0] = *(const v4u*)bl;
                bf.u[1] = *(const v4u*)(bl + 4);
                acc[j] = __builtin_amdgcn_wmma_f32_16x16x32_f16(
                    false, af.h, false, bf.h, (short)0, acc[j], false, false);
            }
            __syncthreads();                     // all reads of `cur` done before overwrite
        }

        if (MODE == 0) {
            float wl = wlag[b * TOPK + kk];
            #pragma unroll
            for (int j = 0; j < 8; ++j) {
                int col = nq * 128 + j * 16 + (lane & 15);
                float bc = bias[col];
                #pragma unroll
                for (int vg = 0; vg < 8; ++vg) {
                    float v = acc[j][vg] + bc;
                    float g = 0.5f * v * (1.0f + erff(v * 0.70710678118654752f));
                    res[j][vg] += wl * g;
                }
            }
        }
    }

    if (MODE == 0) {
        #pragma unroll
        for (int j = 0; j < 8; ++j) {
            int col = nq * 128 + j * 16 + (lane & 15);
            #pragma unroll
            for (int vg = 0; vg < 8; ++vg) {
                int pos = mbase + mt * 16 + ((lane & 16) ? 8 : 0) + vg;
                if (pos < full_len)
                    act2[((size_t)b * SMAX + pos) * CDIM + col] = (_Float16)res[j][vg];
            }
        }
    } else {
        #pragma unroll
        for (int j = 0; j < 8; ++j) {
            int col = nq * 128 + j * 16 + (lane & 15);
            float bc = bias[col];
            #pragma unroll
            for (int vg = 0; vg < 8; ++vg) {
                int pos = mbase + mt * 16 + ((lane & 16) ? 8 : 0) + vg;
                if (pos < TOTAL_L) {
                    size_t o = ((size_t)b * TOTAL_L + pos) * CDIM + col;
                    outp[o] += acc[j][vg] + bc;
                }
            }
        }
    }
}

// ---------------------------------------------------------------- dispatch
extern "C" void kernel_launch(void* const* d_in, const int* in_sizes, int n_in,
                              void* d_out, int out_size, void* d_ws, size_t ws_size,
                              hipStream_t stream) {
    (void)in_sizes; (void)n_in; (void)out_size; (void)ws_size;
    const float* x  = (const float*)d_in[0];
    const float* w1 = (const float*)d_in[1];
    const float* b1 = (const float*)d_in[2];
    const float* w2 = (const float*)d_in[3];
    const float* b2 = (const float*)d_in[4];
    float* out = (float*)d_out;

    char* ws = (char*)d_ws;
    size_t off = 0;
    auto alloc = [&](size_t bytes) -> void* {
        void* p = ws + off;
        off = (off + bytes + 255) & ~(size_t)255;
        return p;
    };
    float* s_buf   = (float*)alloc((size_t)BATCH * SEQ_T * CDIM * 4);
    float* trend   = (float*)alloc((size_t)BATCH * SEQ_T * CDIM * 4);
    float* rbuf    = (float*)alloc((size_t)BATCH * NHEADS * SEQ_T * 4);
    int*   periods = (int*)  alloc((size_t)BATCH * NHEADS * TOPK * 4);
    float* weights = (float*)alloc((size_t)BATCH * NHEADS * TOPK * 4);
    float* wlag    = (float*)alloc((size_t)BATCH * TOPK * 4);
    int*   meta    = (int*)  alloc(16);
    _Float16* act1 = (_Float16*)alloc((size_t)NIMG * SMAX * CDIM * 2);
    _Float16* act2 = (_Float16*)alloc((size_t)BATCH * SMAX * CDIM * 2);
    unsigned int* bp1 = (unsigned int*)alloc((size_t)9 * 16 * 32 * 32 * 8 * 4);
    unsigned int* bp2 = (unsigned int*)alloc((size_t)9 * 16 * 32 * 32 * 8 * 4);

    const size_t conv_smem = (size_t)(2 * 8192 + 2 * 512) * sizeof(unsigned int); // 69,632 B

    init_meta_kernel<<<1, 32, 0, stream>>>(meta);
    trend_kernel<<<(BATCH * CDIM) / 256, 256, 0, stream>>>(x, s_buf, trend, out);
    autocorr_kernel<<<BATCH * NHEADS, SEQ_T, 0, stream>>>(s_buf, rbuf);
    topk_kernel<<<1, BATCH * NHEADS, 0, stream>>>(rbuf, periods, weights, meta);
    finalize_kernel<<<1, 192, 0, stream>>>(weights, wlag, meta);
    roll_kernel<<<((size_t)NIMG * SMAX * 64) / 256, 256, 0, stream>>>(s_buf, periods, meta, act1);
    pack_kernel<<<(9 * 16 * 32 * 32 * 8) / 256, 256, 0, stream>>>(w1, bp1);
    pack_kernel<<<(9 * 16 * 32 * 32 * 8) / 256, 256, 0, stream>>>(w2, bp2);
    conv_kernel<0><<<dim3(SMAX / 32, BATCH), 256, conv_smem, stream>>>(act1, bp1, b1, wlag, meta, act2, nullptr);
    conv_kernel<1><<<dim3(SMAX / 32, BATCH), 256, conv_smem, stream>>>(act2, bp2, b2, nullptr, meta, nullptr, out);
}
